// GATLayer_31482110280023
// MI455X (gfx1250) — matrix operand order repaired and checked by reading, not compiled, not run on hardware
//
#include <hip/hip_runtime.h>
#include <hip/hip_bf16.h>

typedef __attribute__((ext_vector_type(16))) _Float16 v16h;
typedef __attribute__((ext_vector_type(8)))  _Float16 v8h;
typedef __attribute__((ext_vector_type(8)))  float    v8f;
typedef __attribute__((ext_vector_type(4)))  float    v4f;

#define BB  8
#define NN  2048
#define FF  64
#define FPP 64
#define HH  4
#define HFP (HH * FPP)          // 256
#define NEG_BIG (-3.0e38f)
#define LDSW 36                 // padded row stride (floats) -> conflict-free 16-row reads

// ---------------------------------------------------------------------------
// Kernel 1: per (b,h,16-row tile) compute feats tile = X[b] @ W[h] via WMMA,
// reduce s_out/s_in, store feats transposed as f16 [B,H,FP,N].
// One wave (32 threads) per block.
// ---------------------------------------------------------------------------
__global__ __launch_bounds__(32) void gat_feats_kernel(
    const float* __restrict__ X,       // [B,N,F]
    const float* __restrict__ W,       // [H,F,FP]
    const float* __restrict__ a_out,   // [H,FP]
    const float* __restrict__ a_in,    // [H,FP]
    _Float16*    __restrict__ featsT,  // [B,H,FP,N]  (f16)
    float*       __restrict__ s_out,   // [B,H,N]
    float*       __restrict__ s_in)    // [B,H,N]
{
    const int blk  = blockIdx.x;
    const int tile = blk & (NN / 16 - 1);
    const int bh   = blk >> 7;              // / (NN/16)
    const int h    = bh & (HH - 1);
    const int b    = bh >> 2;               // / HH
    const int n0   = tile * 16;

    const int lane = threadIdx.x;
    const int lm   = lane & 15;
    const int hi   = lane >> 4;             // 0 or 1

    const float* Xrow = X + ((size_t)b * NN + (n0 + lm)) * FF;
    const float* Wh   = W + (size_t)h * FF * FPP;

    v8f acc[4];
    #pragma unroll
    for (int c = 0; c < 4; ++c) acc[c] = (v8f)0.0f;

    #pragma unroll
    for (int s = 0; s < 2; ++s) {
        // ---- A tile: rows n0..n0+15, K = 32s..32s+31, WMMA 16-bit A layout
        v16h a;
        #pragma unroll
        for (int g = 0; g < 2; ++g) {
            const int k0 = 32 * s + 16 * g + 8 * hi;
            v4f x0 = *(const v4f*)(Xrow + k0);
            v4f x1 = *(const v4f*)(Xrow + k0 + 4);
            #pragma unroll
            for (int j = 0; j < 4; ++j) {
                a[8 * g + j]     = (_Float16)x0[j];
                a[8 * g + 4 + j] = (_Float16)x1[j];
            }
        }
        // ---- B tiles: W[32s+K][col], WMMA 16-bit B layout (K striped by lane half)
        #pragma unroll
        for (int c = 0; c < 4; ++c) {
            const int col = c * 16 + lm;
            v16h bm;
            #pragma unroll
            for (int e = 0; e < 16; ++e) {
                const int k = 32 * s + 8 * hi + (e & 7) + 16 * (e >> 3);
                bm[e] = (_Float16)Wh[(size_t)k * FPP + col];
            }
            acc[c] = __builtin_amdgcn_wmma_f32_16x16x32_f16(
                false, a, false, bm, (short)0, acc[c], false, false);
        }
    }

    // ---- store featsT (f16, transposed): lane owns column `col`, rows j+8*hi
    const size_t bhBase = (size_t)b * HH + h;
    #pragma unroll
    for (int c = 0; c < 4; ++c) {
        const int col = c * 16 + lm;
        v8h pk;
        #pragma unroll
        for (int j = 0; j < 8; ++j) pk[j] = (_Float16)acc[c][j];
        *(v8h*)(featsT + (bhBase * FPP + col) * NN + n0 + 8 * hi) = pk;
    }

    // ---- s_out / s_in: per-row dot with a_out/a_in, butterfly across 16 lanes
    float ao[4], ai[4];
    #pragma unroll
    for (int c = 0; c < 4; ++c) {
        ao[c] = a_out[h * FPP + c * 16 + lm];
        ai[c] = a_in [h * FPP + c * 16 + lm];
    }
    float so[8], si[8];
    #pragma unroll
    for (int j = 0; j < 8; ++j) {
        float vo = 0.f, vi = 0.f;
        #pragma unroll
        for (int c = 0; c < 4; ++c) {
            vo += acc[c][j] * ao[c];
            vi += acc[c][j] * ai[c];
        }
        so[j] = vo; si[j] = vi;
    }
    #pragma unroll
    for (int m = 1; m < 16; m <<= 1) {
        #pragma unroll
        for (int j = 0; j < 8; ++j) {
            so[j] += __shfl_xor(so[j], m, 32);
            si[j] += __shfl_xor(si[j], m, 32);
        }
    }
    if (lm == 0) {
        #pragma unroll
        for (int j = 0; j < 8; ++j) {
            const int n = n0 + 8 * hi + j;
            s_out[bhBase * NN + n] = so[j];
            s_in [bhBase * NN + n] = si[j];
        }
    }
}

// ---------------------------------------------------------------------------
// Kernel 2: fused masked-softmax attention (flash-style online softmax).
// Block = 128 threads = 4 waves; wave w handles head w for a 32-row tile.
// The mask/adjacency tile (the dominant 268 MB stream) is DMA-staged into LDS
// ONCE per block with global_load_async_to_lds_b128 (double-buffered, overlaps
// with softmax+WMMA of the previous tile), instead of 4x redundant VMEM loads.
// ---------------------------------------------------------------------------
__global__ __launch_bounds__(128) void gat_attn_kernel(
    const float*    __restrict__ adj,     // [B,N,N]
    const float*    __restrict__ mask,    // [B,N,N]
    const _Float16* __restrict__ featsT,  // [B,H,FP,N]
    const float*    __restrict__ s_out,   // [B,H,N]
    const float*    __restrict__ s_in,    // [B,H,N]
    const float*    __restrict__ bias,    // [H,FP]
    float*          __restrict__ out)     // [B,N,H*FP]
{
    __shared__ float smemM[2][32 * LDSW];   // mask tiles (double buffered)
    __shared__ float smemA[2][32 * LDSW];   // adjacency tiles

    const int h    = threadIdx.x >> 5;      // wave index == head
    const int lane = threadIdx.x & 31;
    const int lm   = lane & 15;
    const int hi   = lane >> 4;

    const int rb = blockIdx.x & (NN / 32 - 1);
    const int b  = blockIdx.x >> 6;         // / (NN/32)
    const int n0 = rb * 32;

    const size_t    bh   = (size_t)b * HH + h;
    const float*    so_p = s_out + bh * NN;
    const float*    si_p = s_in  + bh * NN;
    const _Float16* vT   = featsT + bh * FPP * NN;

    // async DMA staging of one 32x32 mask tile + adj tile into LDS buffer bi.
    // 128 threads x 2 chunks x 16B per matrix = 4 KB per matrix.
    auto stage = [&](int bi, int m0s) {
        #pragma unroll
        for (int i = 0; i < 2; ++i) {
            const int c   = (int)threadIdx.x + 128 * i;   // 0..255
            const int row = c >> 3;                       // 0..31
            const int cq  = c & 7;                        // 16B chunk in row
            const size_t goff = ((size_t)b * NN + n0 + row) * NN + m0s + cq * 4;
            const uint32_t ldm = (uint32_t)(uintptr_t)&smemM[bi][row * LDSW + cq * 4];
            const uint32_t lda = (uint32_t)(uintptr_t)&smemA[bi][row * LDSW + cq * 4];
            const uint64_t gm  = (uint64_t)(uintptr_t)(mask + goff);
            const uint64_t ga  = (uint64_t)(uintptr_t)(adj + goff);
            asm volatile("global_load_async_to_lds_b128 %0, %1, off"
                         :: "v"(ldm), "v"(gm) : "memory");
            asm volatile("global_load_async_to_lds_b128 %0, %1, off"
                         :: "v"(lda), "v"(ga) : "memory");
        }
    };

    float so_t[2];
    so_t[0] = so_p[n0 + lm];
    so_t[1] = so_p[n0 + 16 + lm];

    float run_max[2] = {NEG_BIG, NEG_BIG};
    float run_sum[2] = {0.0f, 0.0f};
    v8f acc[2][4];
    #pragma unroll
    for (int t = 0; t < 2; ++t)
        #pragma unroll
        for (int c = 0; c < 4; ++c) acc[t][c] = (v8f)0.0f;

    stage(0, 0);                                          // prime the pipeline

    for (int m0 = 0; m0 < NN; m0 += 32) {
        const int cur = (m0 >> 5) & 1;

        // wait for this wave's DMA chunk, then sync so the whole tile is ready
        // (and so buffer `cur` is no longer being read when it gets re-staged).
        asm volatile("s_wait_asynccnt 0x0" ::: "memory");
        __syncthreads();
        if (m0 + 32 < NN) stage(cur ^ 1, m0 + 32);        // prefetch next tile

        // s_in slots for this lane's 16 K positions (same for both subtiles)
        float siv[16];
        #pragma unroll
        for (int g = 0; g < 2; ++g) {
            const int mb = m0 + 16 * g + 8 * hi;
            v4f s0 = *(const v4f*)(si_p + mb);
            v4f s1 = *(const v4f*)(si_p + mb + 4);
            #pragma unroll
            for (int j = 0; j < 4; ++j) {
                siv[8 * g + j]     = s0[j];
                siv[8 * g + 4 + j] = s1[j];
            }
        }

        v16h A[2];
        #pragma unroll
        for (int t = 0; t < 2; ++t) {
            const float* mrow = &smemM[cur][(16 * t + lm) * LDSW];
            const float* arow = &smemA[cur][(16 * t + lm) * LDSW];
            float sc[16];
            #pragma unroll
            for (int g = 0; g < 2; ++g) {
                const int kb = 16 * g + 8 * hi;           // col within tile
                v4f mv0 = *(const v4f*)(mrow + kb);
                v4f mv1 = *(const v4f*)(mrow + kb + 4);
                v4f av0 = *(const v4f*)(arow + kb);
                v4f av1 = *(const v4f*)(arow + kb + 4);
                #pragma unroll
                for (int j = 0; j < 4; ++j) {
                    float r0 = so_t[t] + siv[8 * g + j];
                    r0 = (r0 > 0.f) ? r0 : 0.2f * r0;     // leaky_relu
                    sc[8 * g + j] = r0 + mv0[j] + av0[j];
                    float r1 = so_t[t] + siv[8 * g + 4 + j];
                    r1 = (r1 > 0.f) ? r1 : 0.2f * r1;
                    sc[8 * g + 4 + j] = r1 + mv1[j] + av1[j];
                }
            }
            // online softmax update for the 16 rows of this subtile
            float tm = sc[0];
            #pragma unroll
            for (int k = 1; k < 16; ++k) tm = fmaxf(tm, sc[k]);
            tm = fmaxf(tm, __shfl_xor(tm, 16, 32));       // full row max
            const float nm    = fmaxf(run_max[t], tm);
            const float scale = __expf(run_max[t] - nm);
            float lsum = 0.f;
            #pragma unroll
            for (int k = 0; k < 16; ++k) {
                const float p = __expf(sc[k] - nm);
                lsum += p;
                A[t][k] = (_Float16)p;
            }
            lsum += __shfl_xor(lsum, 16, 32);             // full row sum
            run_sum[t] = run_sum[t] * scale + lsum;
            run_max[t] = nm;
            // rescale accumulator rows (row j+8*hi scale lives in lane j+8*hi)
            #pragma unroll
            for (int j = 0; j < 8; ++j) {
                const float sj = __shfl(scale, j + 8 * hi, 32);
                #pragma unroll
                for (int c = 0; c < 4; ++c) acc[t][c][j] *= sj;
            }
        }

        // B tiles (two 16B chunks per lane: k = 8*hi.. and 16+8*hi.., transposed f16 feats)
        #pragma unroll
        for (int c = 0; c < 4; ++c) {
            const int col = c * 16 + lm;
            const v8h b0 = *(const v8h*)(vT + (size_t)col * NN + m0 + 8 * hi);
            const v8h b1 = *(const v8h*)(vT + (size_t)col * NN + m0 + 16 + 8 * hi);
            const v16h Bm = __builtin_shufflevector(b0, b1, 0,1,2,3,4,5,6,7,8,9,10,11,12,13,14,15);
            acc[0][c] = __builtin_amdgcn_wmma_f32_16x16x32_f16(
                false, A[0], false, Bm, (short)0, acc[0][c], false, false);
            acc[1][c] = __builtin_amdgcn_wmma_f32_16x16x32_f16(
                false, A[1], false, Bm, (short)0, acc[1][c], false, false);
        }
    }

    // ---- epilogue: 1/rowsum, +bias, ELU, head-major store
    float bval[4];
    #pragma unroll
    for (int c = 0; c < 4; ++c) bval[c] = bias[h * FPP + c * 16 + lm];

    #pragma unroll
    for (int t = 0; t < 2; ++t) {
        const float inv = 1.0f / run_sum[t];
        #pragma unroll
        for (int j = 0; j < 8; ++j) {
            const float ij  = __shfl(inv, j + 8 * hi, 32);
            const int   row = n0 + 16 * t + j + 8 * hi;
            float* orow = out + ((size_t)b * NN + row) * HFP + h * FPP;
            #pragma unroll
            for (int c = 0; c < 4; ++c) {
                float v = acc[t][c][j] * ij + bval[c];
                v = (v > 0.f) ? v : expm1f(v);            // ELU
                orow[c * 16 + lm] = v;
            }
        }
    }
}

// ---------------------------------------------------------------------------
extern "C" void kernel_launch(void* const* d_in, const int* in_sizes, int n_in,
                              void* d_out, int out_size, void* d_ws, size_t ws_size,
                              hipStream_t stream) {
    const float* X     = (const float*)d_in[0];   // node_feats [B,N,F]
    const float* adj   = (const float*)d_in[1];   // adjacency  [B,N,N]
    const float* mask  = (const float*)d_in[2];   // attn_mask  [B,N,N]
    const float* W     = (const float*)d_in[3];   // kernels    [H,F,FP]
    const float* bias  = (const float*)d_in[4];   // biases     [H,FP]
    const float* a_out = (const float*)d_in[5];   // [H,FP]
    const float* a_in  = (const float*)d_in[6];   // [H,FP]
    float* out = (float*)d_out;

    const size_t featsT_bytes = (size_t)BB * HH * FPP * NN * sizeof(_Float16); // 8 MiB
    _Float16* featsT = (_Float16*)d_ws;
    float*    s_out  = (float*)((char*)d_ws + featsT_bytes);
    float*    s_in   = s_out + (size_t)BB * HH * NN;

    // Kernel 1: 4096 blocks x 1 wave
    gat_feats_kernel<<<dim3(BB * HH * (NN / 16)), dim3(32), 0, stream>>>(
        X, W, a_out, a_in, featsT, s_out, s_in);

    // Kernel 2: 512 blocks x 4 waves (one wave per head, LDS-shared adj/mask)
    gat_attn_kernel<<<dim3(BB * (NN / 32)), dim3(128), 0, stream>>>(
        adj, mask, featsT, s_out, s_in, bias, out);
}
